// GraphAttentionLayer_1331439862466
// MI455X (gfx1250) — matrix operand order repaired and checked by reading, not compiled, not run on hardware
//
#include <hip/hip_runtime.h>
#include <hip/hip_bf16.h>

typedef __attribute__((ext_vector_type(16))) _Float16 v16h;
typedef __attribute__((ext_vector_type(8)))  float    v8f;
typedef __attribute__((ext_vector_type(8)))  unsigned v8u;

#define ALPHA 0.2f
#define MASK_FILL -9000000000000000.0f
#define NN 2048
#define FF 256
#define BS 4

// pack two f32 -> packed f16x2 in a u32 (lowers to v_cvt_pk_f16_f32)
__device__ __forceinline__ unsigned pkh2(float a, float b) {
    union { _Float16 h[2]; unsigned u; } p;
    p.h[0] = (_Float16)a;
    p.h[1] = (_Float16)b;
    return p.u;
}

// -------------------------------------------------------------------------
// Kernel 1: Wh = h @ W  (per batch, M=2048, N=256, K=256).
// Block = 8 waves; wave owns one 16-row i-tile, accumulates all 16 o-tiles.
// LDS: h tile as f16 (A pairs adjacent -> ds_load_b32), W chunk pair-packed
// ({W[2q,o],W[2q+1,o]} per u32 -> B fragment = 8x ds_load_b32 + bit_cast).
// Outputs: Wh (f32, row-major) and Whp (f16 pair-major: word (b,q= j/2, o)).
// Grid: (16 i-tile-groups, 4 batches), 256 threads.
// -------------------------------------------------------------------------
__global__ void __launch_bounds__(256)
gat_gemm_wh(const float* __restrict__ h,
            const float* __restrict__ W,
            float* __restrict__ Wh,
            unsigned* __restrict__ Whp) {
    __shared__ __align__(16) _Float16 sh16[128 * 32];  // 8 KB: h tile (f16)
    __shared__ __align__(16) unsigned sWu[16 * 256];   // 16 KB: W chunk, pair-packed

    const int tid  = threadIdx.x;
    const int lane = tid & 31;
    const int wave = tid >> 5;
    const int b    = blockIdx.y;
    const int rowbase = blockIdx.x * 128;
    const int it   = blockIdx.x * 8 + wave;
    const int half = lane >> 4;
    const int l15  = lane & 15;

    unsigned* shU = (unsigned*)sh16;
    v8f acc[16] = {};

    for (int k0 = 0; k0 < FF; k0 += 32) {
        __syncthreads();
        // stage h tile: 128 rows x 32 k, float4 loads + pk-f16 convert
        for (int e = tid; e < 1024; e += 256) {          // 1024 float4 = 128x32
            int r = e >> 3, c4 = e & 7;
            float4 x = ((const float4*)(h + ((b * NN) + rowbase + r) * FF + k0))[c4];
            shU[e * 2]     = pkh2(x.x, x.y);
            shU[e * 2 + 1] = pkh2(x.z, x.w);
        }
        // stage W chunk pair-packed: word q*256+c = {W[k0+2q,c], W[k0+2q+1,c]}
        for (int e = tid; e < 16 * 256; e += 256) {
            int q = e >> 8, c = e & 255;
            sWu[e] = pkh2(W[(k0 + 2 * q) * FF + c], W[(k0 + 2 * q + 1) * FF + c]);
        }
        __syncthreads();

        // A fragment: 8 x ds_load_b32, zero repacking
        v8u afu;
        const int rl = wave * 16 + l15;
#pragma unroll
        for (int v = 0; v < 8; ++v)
            afu[v] = shU[rl * 16 + ((v >> 2) << 3) + (half << 2) + (v & 3)];
        v16h af = __builtin_bit_cast(v16h, afu);

#pragma unroll
        for (int ot = 0; ot < 16; ++ot) {
            v8u bfu;
            const int o = ot * 16 + l15;
#pragma unroll
            for (int v = 0; v < 8; ++v)
                bfu[v] = sWu[(((v >> 2) << 3) + (half << 2) + (v & 3)) * 256 + o];
            v16h bf = __builtin_bit_cast(v16h, bfu);
            acc[ot] = __builtin_amdgcn_wmma_f32_16x16x32_f16(
                false, af, false, bf, (short)0, acc[ot], false, false);
        }
    }

#pragma unroll
    for (int ot = 0; ot < 16; ++ot) {
        const int c = ot * 16 + l15;
#pragma unroll
        for (int v = 0; v < 8; v += 2) {
            int r = it * 16 + (half << 3) + v;           // v even -> r,r+1 pair
            Wh[((b * NN) + r) * FF + c]     = acc[ot][v];
            Wh[((b * NN) + r + 1) * FF + c] = acc[ot][v + 1];
            // pair-major f16: word index ((b*1024 + r/2)*256 + c)
            Whp[((b * (NN / 2)) + (r >> 1)) * FF + c] = pkh2(acc[ot][v], acc[ot][v + 1]);
        }
    }
}

// -------------------------------------------------------------------------
// Kernel 2: f1[b,i] = Wh[b,i,:].a1 ; f2[b,i] = Wh[b,i,:].a2
// One wave per row, shuffle reduction (wave32).
// -------------------------------------------------------------------------
__global__ void gat_f1f2(const float* __restrict__ Wh,
                         const float* __restrict__ a,
                         float* __restrict__ f1,
                         float* __restrict__ f2) {
    const int lane = threadIdx.x & 31;
    const int w    = blockIdx.x * 8 + (threadIdx.x >> 5);  // 0..8191
    const int b    = w >> 11;
    const int i    = w & (NN - 1);
    const float* wr = Wh + ((b * NN) + i) * FF;

    float s1 = 0.f, s2 = 0.f;
#pragma unroll
    for (int t = 0; t < FF / 32; ++t) {
        int k = lane + 32 * t;
        float wv = wr[k];
        s1 += wv * a[k];
        s2 += wv * a[FF + k];
    }
#pragma unroll
    for (int off = 16; off > 0; off >>= 1) {
        s1 += __shfl_down(s1, off, 32);
        s2 += __shfl_down(s2, off, 32);
    }
    if (lane == 0) {
        f1[b * NN + i] = s1;
        f2[b * NN + i] = s2;
    }
}

// -------------------------------------------------------------------------
// Kernel 3: column-wise softmax stats (reference softmax is over axis=1, i.e.
// over i for each column j). Online max/sum; mask reads coalesced across j.
// Grid: (8 j-groups, 4 batches), 256 threads. Writes colmax and 1/colsum.
// -------------------------------------------------------------------------
__global__ void gat_colstats(const int* __restrict__ mask,
                             const float* __restrict__ f1,
                             const float* __restrict__ f2,
                             float* __restrict__ cm,
                             float* __restrict__ crcp) {
    __shared__ float sf1[256];
    const int tid = threadIdx.x;
    const int b   = blockIdx.y;
    const int j   = blockIdx.x * 256 + tid;

    const float f2v = f2[b * NN + j];
    float m = -INFINITY, s = 0.f;

    for (int c = 0; c < NN / 256; ++c) {
        __syncthreads();
        sf1[tid] = f1[b * NN + c * 256 + tid];
        __syncthreads();
        int ibase = ((b * NN) + c * 256) * NN + j;
#pragma unroll 4
        for (int ii = 0; ii < 256; ++ii) {
            int mv  = mask[ibase + ii * NN];
            float e = sf1[ii] + f2v;
            e = (e >= 0.f) ? e : ALPHA * e;
            float att = (mv > 0) ? e : MASK_FILL;
            float nm  = fmaxf(m, att);
            s = s * __expf(m - nm) + __expf(att - nm);
            m = nm;
        }
    }
    cm[b * NN + j]   = m;
    crcp[b * NN + j] = 1.f / s;
}

// -------------------------------------------------------------------------
// Kernel 4: out = elu( softmaxP @ Wh ).
// Mask tile staged with int4 (b128) copies; Whp chunk is a contiguous 16KB
// b128 block copy (pair-major produced by kernel 1). P fragment computed
// ONCE per (i,j); B fragments are 8x ds_load_b32 + bit_cast. 16 WMMAs per
// A fragment. elu epilogue.
// Grid: (16 i-tile-groups, 4 batches), 256 threads.
// -------------------------------------------------------------------------
__global__ void __launch_bounds__(256)
gat_attn_gemm(const int* __restrict__ mask,
              const float* __restrict__ f1,
              const float* __restrict__ f2,
              const float* __restrict__ cm,
              const float* __restrict__ crcp,
              const unsigned* __restrict__ Whp,
              float* __restrict__ out) {
    __shared__ float sf2[NN];                          // 8 KB
    __shared__ float scm[NN];                          // 8 KB
    __shared__ float srcp[NN];                         // 8 KB
    __shared__ __align__(16) int      smask[128 * 32]; // 16 KB
    __shared__ __align__(16) unsigned sWu[16 * 256];   // 16 KB, pair-packed

    const int tid  = threadIdx.x;
    const int lane = tid & 31;
    const int wave = tid >> 5;
    const int b    = blockIdx.y;
    const int rowbase = blockIdx.x * 128;
    const int it   = blockIdx.x * 8 + wave;
    const int half = lane >> 4;
    const int l15  = lane & 15;

    for (int idx = tid; idx < NN; idx += 256) {
        sf2[idx]  = f2[b * NN + idx];
        scm[idx]  = cm[b * NN + idx];
        srcp[idx] = crcp[b * NN + idx];
    }

    const float f1i = f1[b * NN + it * 16 + l15];
    v8f acc[16] = {};

    for (int j0 = 0; j0 < NN; j0 += 32) {
        __syncthreads();
        // stage mask tile: 128 rows x 32 j as int4 (b128), coalesced
        for (int e = tid; e < 1024; e += 256) {          // 1024 int4
            int r = e >> 3, c4 = e & 7;
            ((int4*)smask)[e] =
                ((const int4*)(mask + ((b * NN) + rowbase + r) * NN + j0))[c4];
        }
        // stage Whp chunk: contiguous 16KB block copy (b128)
        {
            const uint4* src = (const uint4*)(Whp + ((b * (NN / 2)) + (j0 >> 1)) * FF);
            uint4* dst = (uint4*)sWu;
            for (int e = tid; e < 1024; e += 256)        // 1024 uint4
                dst[e] = src[e];
        }
        __syncthreads();

        // prefetch next chunk's mask slice (gfx1250 global_prefetch_b8)
        if (j0 + 32 < NN)
            __builtin_prefetch(&mask[((b * NN) + rowbase + (tid >> 5)) * NN +
                                     j0 + 32 + (tid & 31)], 0, 1);

        // Build the softmaxed-P A-fragment ONCE per chunk
        v8u afu;
        const int rl = wave * 16 + l15;
#pragma unroll
        for (int v = 0; v < 8; ++v) {
            int jl = ((v >> 2) << 4) + (half << 3) + ((v & 3) << 1);  // even
            int2 mp = ((const int2*)smask)[(rl * 32 + jl) >> 1];       // ds_load_b64
            float p01[2];
#pragma unroll
            for (int s = 0; s < 2; ++s) {
                int j   = j0 + jl + s;
                int mv  = (s == 0) ? mp.x : mp.y;
                float e = f1i + sf2[j];
                e = (e >= 0.f) ? e : ALPHA * e;
                float att = (mv > 0) ? e : MASK_FILL;
                p01[s] = __expf(att - scm[j]) * srcp[j];
            }
            afu[v] = pkh2(p01[0], p01[1]);
        }
        v16h af = __builtin_bit_cast(v16h, afu);

        // 16 WMMAs reuse the A fragment
#pragma unroll
        for (int ot = 0; ot < 16; ++ot) {
            v8u bfu;
            const int o = ot * 16 + l15;
#pragma unroll
            for (int v = 0; v < 8; ++v)
                bfu[v] = sWu[(((v >> 2) << 3) + (half << 2) + (v & 3)) * 256 + o];
            v16h bf = __builtin_bit_cast(v16h, bfu);
            acc[ot] = __builtin_amdgcn_wmma_f32_16x16x32_f16(
                false, af, false, bf, (short)0, acc[ot], false, false);
        }
    }

#pragma unroll
    for (int ot = 0; ot < 16; ++ot) {
        const int c = ot * 16 + l15;
#pragma unroll
        for (int v = 0; v < 8; ++v) {
            int r   = it * 16 + (half << 3) + v;
            float x = acc[ot][v];
            // elu (alpha=1): x>0 ? x : exp(x)-1
            out[((b * NN) + r) * FF + c] = (x > 0.f) ? x : expm1f(x);
        }
    }
}

// -------------------------------------------------------------------------
// Launch
// -------------------------------------------------------------------------
extern "C" void kernel_launch(void* const* d_in, const int* in_sizes, int n_in,
                              void* d_out, int out_size, void* d_ws, size_t ws_size,
                              hipStream_t stream) {
    const float* h    = (const float*)d_in[0];   // (4,2048,256) f32
    const int*   mask = (const int*)d_in[1];     // (4,2048,2048) i32
    const float* W    = (const float*)d_in[2];   // (256,256) f32
    const float* a    = (const float*)d_in[3];   // (512,1) f32
    float* out = (float*)d_out;                  // (4,2048,256) f32

    char* ws = (char*)d_ws;
    float*    Wh   = (float*)ws;                                   // 8 MB
    unsigned* Whp  = (unsigned*)(ws + (size_t)8 * 1024 * 1024);    // 4 MB (f16 pairs)
    float*    f1   = (float*)(ws + (size_t)12 * 1024 * 1024);      // 32 KB
    float*    f2   = f1 + BS * NN;
    float*    cmx  = f2 + BS * NN;
    float*    crcp = cmx + BS * NN;

    dim3 gGemm(16, 4, 1);   // 16 i-tile-groups x 4 batches, 8 waves/block
    gat_gemm_wh<<<gGemm, 256, 0, stream>>>(h, W, Wh, Whp);

    gat_f1f2<<<dim3(1024, 1, 1), 256, 0, stream>>>(Wh, a, f1, f2);

    gat_colstats<<<dim3(8, 4, 1), 256, 0, stream>>>(mask, f1, f2, cmx, crcp);

    gat_attn_gemm<<<gGemm, 256, 0, stream>>>(mask, f1, f2, cmx, crcp, Whp, out);
}
